// TransformerBlock_33028298506755
// MI455X (gfx1250) — matrix operand order repaired and checked by reading, not compiled, not run on hardware
//
#include <hip/hip_runtime.h>

typedef unsigned int uint;
typedef unsigned short ushort;
typedef __attribute__((ext_vector_type(16))) __bf16 v16bf;
typedef __attribute__((ext_vector_type(8)))  float  v8f;
typedef __attribute__((ext_vector_type(4)))  uint   u32x4;
typedef __attribute__((ext_vector_type(8)))  uint   u32x8;

#define D_MODEL 1024
#define N_HEADS 16
#define D_K     64
#define D_FF    4096
#define BB      2
#define TT      2048
#define MTOT    (BB*TT)   // 4096 rows

union AFrag { uint4 u[2]; v16bf v; };
union Row32 { uint4 q[4]; ushort s[32]; };

__device__ __forceinline__ ushort f2bf(float f) {
    uint u = __float_as_uint(f);
    u += 0x7fffu + ((u >> 16) & 1u);        // round-to-nearest-even
    return (ushort)(u >> 16);
}

__device__ __forceinline__ float rmax16(float v) {
    #pragma unroll
    for (int m = 1; m < 16; m <<= 1) v = fmaxf(v, __shfl_xor(v, m, 16));
    return v;
}
__device__ __forceinline__ float rsum16(float v) {
    #pragma unroll
    for (int m = 1; m < 16; m <<= 1) v += __shfl_xor(v, m, 16);
    return v;
}

// ---------------------------------------------------------------------------
// TDM: async 2-D tile load global->LDS (wave-level DMA, TENSORcnt-tracked).
// Tile = tile_d1 rows x 32 bf16, row stride in memory = stride0 elements.
// LDS rows are padded to 40 elements via TDM pad (16 DWORDs data + 4 DWORDs pad).
// Descriptor packing per CDNA5 ISA §8.3-8.6.
// ---------------------------------------------------------------------------
__device__ __forceinline__ void tdm_load_tile32(
    uint lds_addr, const ushort* gptr,
    uint tile_d1, uint tensor_d0, uint tensor_d1, uint stride0)
{
    unsigned long long ga = (unsigned long long)(size_t)gptr;
    u32x4 g0;
    g0[0] = 1u;                                              // count=1, no gather
    g0[1] = lds_addr;                                        // LDS byte address
    g0[2] = (uint)ga;                                        // global_addr[31:0]
    g0[3] = ((uint)(ga >> 32) & 0x01FFFFFFu) | 0x80000000u;  // addr[56:32] | type=2
    u32x8 g1;
    g1[0] = (1u << 16)        // data_size = 2 bytes
          | (1u << 20)        // pad_enable
          | (3u << 22)        // pad_interval: 16 DWORDs (=64B = one 32-elem row)
          | (3u << 25);       // pad_amount: 4 DWORDs (=16B = 8 elems -> stride 40)
    g1[1] = (tensor_d0 & 0xFFFFu) << 16;                       // tensor_dim0[15:0]
    g1[2] = (tensor_d0 >> 16) | ((tensor_d1 & 0xFFFFu) << 16); // dim0 hi / dim1 lo
    g1[3] = (tensor_d1 >> 16) | (32u << 16);                   // dim1 hi / tile_dim0=32
    g1[4] = tile_d1;                                           // tile_dim1, tile_dim2=0
    g1[5] = stride0;                                           // dim0_stride[31:0]
    g1[6] = 0u;                                                // stride hi, dim1_stride lo
    g1[7] = 0u;
    u32x4 g2 = (u32x4)(0u);
    u32x4 g3 = (u32x4)(0u);
    asm volatile("tensor_load_to_lds %0, %1, %2, %3"
                 :: "s"(g0), "s"(g1), "s"(g2), "s"(g3) : "memory");
}

// ---------------------------------------------------------------------------
// Weight convert + transpose: Wt[n][k] = bf16(W[k][n]).  W is [K][N] row-major.
// ---------------------------------------------------------------------------
__global__ __launch_bounds__(256) void transpose_bf16_kernel(
    const float* __restrict__ W, ushort* __restrict__ Wt, int K, int N)
{
    __shared__ float t[32][33];
    const int n0 = blockIdx.x * 32, k0 = blockIdx.y * 32;
    const int tx = threadIdx.x, ty = threadIdx.y;   // (32, 8)
    #pragma unroll
    for (int j = 0; j < 4; ++j) {
        int r = ty + j * 8;
        t[r][tx] = W[(size_t)(k0 + r) * N + n0 + tx];
    }
    __syncthreads();
    #pragma unroll
    for (int j = 0; j < 4; ++j) {
        int r = ty + j * 8;
        Wt[(size_t)(n0 + r) * K + k0 + tx] = f2bf(t[tx][r]);
    }
}

// ---------------------------------------------------------------------------
// LayerNorm: one block per row of 1024; out is bf16.
// ---------------------------------------------------------------------------
__global__ __launch_bounds__(256) void ln_kernel(
    const float* __restrict__ x, const float* __restrict__ g,
    const float* __restrict__ b, ushort* __restrict__ out)
{
    __shared__ float s1[256], s2[256];
    __shared__ float mu_s, ri_s;
    const int row = blockIdx.x, tid = threadIdx.x;
    const float* xr = x + (size_t)row * D_MODEL;
    float v[4]; float sum = 0.f, sq = 0.f;
    #pragma unroll
    for (int i = 0; i < 4; ++i) {
        v[i] = xr[tid + 256 * i];
        sum += v[i]; sq += v[i] * v[i];
    }
    s1[tid] = sum; s2[tid] = sq;
    __syncthreads();
    for (int st = 128; st > 0; st >>= 1) {
        if (tid < st) { s1[tid] += s1[tid + st]; s2[tid] += s2[tid + st]; }
        __syncthreads();
    }
    if (tid == 0) {
        float mu = s1[0] * (1.f / D_MODEL);
        float var = s2[0] * (1.f / D_MODEL) - mu * mu;
        mu_s = mu; ri_s = rsqrtf(var + 1e-5f);
    }
    __syncthreads();
    const float mu = mu_s, ri = ri_s;
    ushort* o = out + (size_t)row * D_MODEL;
    #pragma unroll
    for (int i = 0; i < 4; ++i) {
        int c = tid + 256 * i;
        o[c] = f2bf((v[i] - mu) * ri * g[c] + b[c]);
    }
}

// ---------------------------------------------------------------------------
// Tiled WMMA GEMM with TDM double buffering:
//   C[M,N] = A[M,K](bf16) * Bt[N,K](bf16)^T + bias
//   mode 0: out = bf16(acc+bias);  mode 2: out = bf16(gelu(acc+bias))
//   mode 1/3: out = f32(resid + acc + bias)
// Block tile 128x128, 8 waves (2M x 4N), wave tile 64x32, K-step 32.
// Wave 0 drives the TDM; tile k+1 streams in while all waves compute tile k.
// ---------------------------------------------------------------------------
__global__ __launch_bounds__(256) void gemm_bf16_kernel(
    const ushort* __restrict__ A, const ushort* __restrict__ Bt,
    const float* __restrict__ bias, const float* __restrict__ resid,
    void* __restrict__ outp, int M, int N, int K, int mode)
{
    __shared__ ushort As[2][128 * 40];   // rows padded to 40 elems by TDM
    __shared__ ushort Bs[2][128 * 40];

    const int tid = threadIdx.x;
    const int lane = tid & 31, w = tid >> 5;
    const int lane16 = lane & 15, half = lane >> 4;
    const int m0 = blockIdx.y * 128, n0 = blockIdx.x * 128;
    const int wm = (w & 1) * 64, wn = (w >> 1) * 32;

    v8f acc[4][2];
    #pragma unroll
    for (int i = 0; i < 4; ++i)
        #pragma unroll
        for (int j = 0; j < 2; ++j)
            #pragma unroll
            for (int e = 0; e < 8; ++e) acc[i][j][e] = 0.f;

    const uint lA0 = (uint)(size_t)&As[0][0], lA1 = (uint)(size_t)&As[1][0];
    const uint lB0 = (uint)(size_t)&Bs[0][0], lB1 = (uint)(size_t)&Bs[1][0];
    const ushort* gA = A + (size_t)m0 * K;
    const ushort* gB = Bt + (size_t)n0 * K;

    const int nt = K >> 5;           // K/32 tiles
    if (w == 0) {                    // prologue: tile 0 into buffer 0
        tdm_load_tile32(lA0, gA, 128, (uint)K, (uint)M, (uint)K);
        tdm_load_tile32(lB0, gB, 128, (uint)K, (uint)N, (uint)K);
    }

    for (int kt = 0; kt < nt; ++kt) {
        if (w == 0) __builtin_amdgcn_s_wait_tensorcnt(0);   // tile kt arrived
        __syncthreads();                                    // publish tile kt
        if (w == 0 && kt + 1 < nt) {                        // stream tile kt+1
            const uint la = ((kt + 1) & 1) ? lA1 : lA0;
            const uint lb = ((kt + 1) & 1) ? lB1 : lB0;
            tdm_load_tile32(la, gA + (kt + 1) * 32, 128, (uint)K, (uint)M, (uint)K);
            tdm_load_tile32(lb, gB + (kt + 1) * 32, 128, (uint)K, (uint)N, (uint)K);
        }
        const ushort* as = As[kt & 1];
        const ushort* bs = Bs[kt & 1];

        AFrag af[4], bf[2];
        #pragma unroll
        for (int i = 0; i < 4; ++i) {
            int m = wm + i * 16 + lane16;
            af[i].u[0] = *(const uint4*)&as[m * 40 + 8 * half];       // k = 8h..
            af[i].u[1] = *(const uint4*)&as[m * 40 + 16 + 8 * half];  // k = 16+8h..
        }
        #pragma unroll
        for (int j = 0; j < 2; ++j) {
            int n = wn + j * 16 + lane16;
            bf[j].u[0] = *(const uint4*)&bs[n * 40 + 8 * half];       // k = 8h..
            bf[j].u[1] = *(const uint4*)&bs[n * 40 + 16 + 8 * half];
        }
        #pragma unroll
        for (int i = 0; i < 4; ++i)
            #pragma unroll
            for (int j = 0; j < 2; ++j)
                acc[i][j] = __builtin_amdgcn_wmma_f32_16x16x32_bf16(
                    false, af[i].v, false, bf[j].v, (short)0, acc[i][j], false, false);
    }

    const bool to_f32 = (mode == 1) || (mode == 3);
    #pragma unroll
    for (int i = 0; i < 4; ++i) {
        #pragma unroll
        for (int j = 0; j < 2; ++j) {
            const int gn = n0 + wn + j * 16 + lane16;
            const float bv = bias[gn];
            #pragma unroll
            for (int r = 0; r < 8; ++r) {
                const int gm = m0 + wm + i * 16 + r + 8 * half;
                const size_t idx = (size_t)gm * N + gn;
                float v = acc[i][j][r] + bv;
                if (to_f32) {
                    ((float*)outp)[idx] = resid[idx] + v;
                } else {
                    if (mode == 2) v = 0.5f * v * (1.f + erff(v * 0.70710678118654752f));
                    ((ushort*)outp)[idx] = f2bf(v);
                }
            }
        }
    }
}

// ---------------------------------------------------------------------------
// Causal flash attention. grid (T/128, H, B), 8 waves, one wave = 16 q rows.
// Q,K,V,O are bf16 [B,T,D_MODEL] with head h at cols h*64..h*64+63.
// ---------------------------------------------------------------------------
__global__ __launch_bounds__(256) void attn_kernel(
    const ushort* __restrict__ Q, const ushort* __restrict__ Kb,
    const ushort* __restrict__ Vb, ushort* __restrict__ O)
{
    __shared__ ushort Vt[8][64 * 40];   // per wave: Vt[d][kv], rows padded to 40
    __shared__ ushort Pl[8][16 * 40];   // per wave: P[m][kv]

    const int tid = threadIdx.x;
    const int lane = tid & 31, w = tid >> 5;
    const int lane16 = lane & 15, half = lane >> 4;
    const int qbase = blockIdx.x * 128 + w * 16;
    const int head = blockIdx.y, batch = blockIdx.z;
    const size_t hoff = (size_t)head * D_K;
    const size_t bt = (size_t)batch * TT;

    AFrag qa[2];   // A-layout: row m = lane16, k = d
    {
        const ushort* qrow = Q + (bt + qbase + lane16) * D_MODEL + hoff;
        #pragma unroll
        for (int c = 0; c < 2; ++c) {
            qa[c].u[0] = *(const uint4*)(qrow + c * 32 + 8 * half);
            qa[c].u[1] = *(const uint4*)(qrow + c * 32 + 16 + 8 * half);
        }
    }

    float mrow[8], lrow[8];
    v8f oacc[4];
    #pragma unroll
    for (int r = 0; r < 8; ++r) { mrow[r] = -3.0e38f; lrow[r] = 0.f; }
    #pragma unroll
    for (int d = 0; d < 4; ++d)
        #pragma unroll
        for (int r = 0; r < 8; ++r) oacc[d][r] = 0.f;

    const int kvmax = qbase + 15;
    for (int kv0 = 0; kv0 <= kvmax; kv0 += 32) {
        // ---- S = (Q K^T) * 1/8, causal-masked ----
        float sm[2][8];
        #pragma unroll
        for (int s = 0; s < 2; ++s) {
            const ushort* krow = Kb + (bt + kv0 + s * 16 + lane16) * D_MODEL + hoff;
            if (kv0 + 32 <= kvmax) __builtin_prefetch(krow + 32 * D_MODEL);
            AFrag kf0, kf1;                       // B-frags: n = kv row, k = d
            kf0.u[0] = *(const uint4*)(krow + 8 * half);
            kf0.u[1] = *(const uint4*)(krow + 16 + 8 * half);
            kf1.u[0] = *(const uint4*)(krow + 32 + 8 * half);
            kf1.u[1] = *(const uint4*)(krow + 32 + 16 + 8 * half);
            v8f sc;
            #pragma unroll
            for (int e = 0; e < 8; ++e) sc[e] = 0.f;
            sc = __builtin_amdgcn_wmma_f32_16x16x32_bf16(false, qa[0].v, false, kf0.v, (short)0, sc, false, false);
            sc = __builtin_amdgcn_wmma_f32_16x16x32_bf16(false, qa[1].v, false, kf1.v, (short)0, sc, false, false);
            #pragma unroll
            for (int r = 0; r < 8; ++r) {
                int qq = qbase + r + 8 * half;
                int kk = kv0 + s * 16 + lane16;
                sm[s][r] = (kk <= qq) ? sc[r] * 0.125f : -3.0e38f;
            }
        }
        // ---- online softmax update; stage P (bf16) into LDS ----
        #pragma unroll
        for (int r = 0; r < 8; ++r) {
            float rm = rmax16(fmaxf(sm[0][r], sm[1][r]));
            float mn = fmaxf(mrow[r], rm);
            float corr = __expf(mrow[r] - mn);
            mrow[r] = mn;
            float p0 = __expf(sm[0][r] - mn);
            float p1 = __expf(sm[1][r] - mn);
            lrow[r] = lrow[r] * corr + rsum16(p0 + p1);
            #pragma unroll
            for (int d = 0; d < 4; ++d) oacc[d][r] *= corr;
            int m = r + 8 * half;
            Pl[w][m * 40 + lane16]      = f2bf(p0);
            Pl[w][m * 40 + 16 + lane16] = f2bf(p1);
        }
        // ---- stage V transposed: Vt[d][kv] via packed kv-pair b32 stores ----
        {
            const int l = lane & 15, dh = lane >> 4;
            const ushort* v0 = Vb + (bt + kv0 + 2 * l) * D_MODEL + hoff + dh * 32;
            Row32 r0, r1;
            #pragma unroll
            for (int i = 0; i < 4; ++i) {
                r0.q[i] = ((const uint4*)v0)[i];
                r1.q[i] = ((const uint4*)(v0 + D_MODEL))[i];
            }
            ushort* vt = &Vt[w][0];
            #pragma unroll
            for (int d = 0; d < 32; ++d) {
                uint val = (uint)r0.s[d] | ((uint)r1.s[d] << 16);
                *(uint*)&vt[(dh * 32 + d) * 40 + 2 * l] = val;
            }
        }
        asm volatile("s_wait_dscnt 0" ::: "memory");
        // ---- O += P * V ----
        AFrag pf;
        pf.u[0] = *(const uint4*)&Pl[w][lane16 * 40 + 8 * half];
        pf.u[1] = *(const uint4*)&Pl[w][lane16 * 40 + 16 + 8 * half];
        #pragma unroll
        for (int d = 0; d < 4; ++d) {
            AFrag vf;   // B-frag: n = output d col, k = kv
            vf.u[0] = *(const uint4*)&Vt[w][(d * 16 + lane16) * 40 + 8 * half];
            vf.u[1] = *(const uint4*)&Vt[w][(d * 16 + lane16) * 40 + 16 + 8 * half];
            oacc[d] = __builtin_amdgcn_wmma_f32_16x16x32_bf16(
                false, pf.v, false, vf.v, (short)0, oacc[d], false, false);
        }
    }
    // ---- normalize and write O (bf16) ----
    #pragma unroll
    for (int d = 0; d < 4; ++d) {
        #pragma unroll
        for (int r = 0; r < 8; ++r) {
            float val = oacc[d][r] / lrow[r];
            size_t idx = (bt + qbase + r + 8 * half) * D_MODEL + hoff + d * 16 + lane16;
            O[idx] = f2bf(val);
        }
    }
}

// ---------------------------------------------------------------------------
// Host-side launcher
// ---------------------------------------------------------------------------
extern "C" void kernel_launch(void* const* d_in, const int* in_sizes, int n_in,
                              void* d_out, int out_size, void* d_ws, size_t ws_size,
                              hipStream_t stream) {
    (void)in_sizes; (void)n_in; (void)out_size; (void)ws_size;
    const float* x     = (const float*)d_in[0];
    const float* ln1_g = (const float*)d_in[2];
    const float* ln1_b = (const float*)d_in[3];
    const float* wq = (const float*)d_in[4];  const float* bq = (const float*)d_in[5];
    const float* wk = (const float*)d_in[6];  const float* bk = (const float*)d_in[7];
    const float* wv = (const float*)d_in[8];  const float* bv = (const float*)d_in[9];
    const float* wo = (const float*)d_in[10]; const float* bo = (const float*)d_in[11];
    const float* ln2_g = (const float*)d_in[12];
    const float* ln2_b = (const float*)d_in[13];
    const float* w1 = (const float*)d_in[14]; const float* b1 = (const float*)d_in[15];
    const float* w2 = (const float*)d_in[16]; const float* b2 = (const float*)d_in[17];

    char* p = (char*)d_ws;
    const size_t DD  = (size_t)D_MODEL * D_MODEL;
    const size_t DF  = (size_t)D_MODEL * D_FF;
    const size_t MD  = (size_t)MTOT * D_MODEL;
    const size_t MF  = (size_t)MTOT * D_FF;
    ushort* wqT = (ushort*)p; p += DD * 2;
    ushort* wkT = (ushort*)p; p += DD * 2;
    ushort* wvT = (ushort*)p; p += DD * 2;
    ushort* woT = (ushort*)p; p += DD * 2;
    ushort* w1T = (ushort*)p; p += DF * 2;
    ushort* w2T = (ushort*)p; p += DF * 2;
    ushort* hb  = (ushort*)p; p += MD * 2;
    ushort* qb  = (ushort*)p; p += MD * 2;
    ushort* kb  = (ushort*)p; p += MD * 2;
    ushort* vbuf= (ushort*)p; p += MD * 2;
    ushort* ob  = (ushort*)p; p += MD * 2;
    float*  x1  = (float*)p;  p += MD * 4;
    ushort* h2b = (ushort*)p; p += MD * 2;
    ushort* f1b = (ushort*)p; p += MF * 2;

    const dim3 tb(32, 8);
    transpose_bf16_kernel<<<dim3(D_MODEL/32, D_MODEL/32), tb, 0, stream>>>(wq, wqT, D_MODEL, D_MODEL);
    transpose_bf16_kernel<<<dim3(D_MODEL/32, D_MODEL/32), tb, 0, stream>>>(wk, wkT, D_MODEL, D_MODEL);
    transpose_bf16_kernel<<<dim3(D_MODEL/32, D_MODEL/32), tb, 0, stream>>>(wv, wvT, D_MODEL, D_MODEL);
    transpose_bf16_kernel<<<dim3(D_MODEL/32, D_MODEL/32), tb, 0, stream>>>(wo, woT, D_MODEL, D_MODEL);
    transpose_bf16_kernel<<<dim3(D_FF/32,    D_MODEL/32), tb, 0, stream>>>(w1, w1T, D_MODEL, D_FF);
    transpose_bf16_kernel<<<dim3(D_MODEL/32, D_FF/32),    tb, 0, stream>>>(w2, w2T, D_FF, D_MODEL);

    ln_kernel<<<MTOT, 256, 0, stream>>>(x, ln1_g, ln1_b, hb);

    const dim3 gD(D_MODEL/128, MTOT/128);
    gemm_bf16_kernel<<<gD, 256, 0, stream>>>(hb, wqT, bq, nullptr, qb,   MTOT, D_MODEL, D_MODEL, 0);
    gemm_bf16_kernel<<<gD, 256, 0, stream>>>(hb, wkT, bk, nullptr, kb,   MTOT, D_MODEL, D_MODEL, 0);
    gemm_bf16_kernel<<<gD, 256, 0, stream>>>(hb, wvT, bv, nullptr, vbuf, MTOT, D_MODEL, D_MODEL, 0);

    attn_kernel<<<dim3(TT/128, N_HEADS, BB), 256, 0, stream>>>(qb, kb, vbuf, ob);

    gemm_bf16_kernel<<<gD, 256, 0, stream>>>(ob, woT, bo, x, x1, MTOT, D_MODEL, D_MODEL, 1);

    ln_kernel<<<MTOT, 256, 0, stream>>>(x1, ln2_g, ln2_b, h2b);

    gemm_bf16_kernel<<<dim3(D_FF/128, MTOT/128), 256, 0, stream>>>(h2b, w1T, b1, nullptr, f1b, MTOT, D_FF, D_MODEL, 2);

    gemm_bf16_kernel<<<gD, 256, 0, stream>>>(f1b, w2T, b2, x1, (float*)d_out, MTOT, D_MODEL, D_FF, 3);
}
